// CPModel_46497315946702
// MI455X (gfx1250) — matrix operand order repaired and checked by reading, not compiled, not run on hardware
//
#include <hip/hip_runtime.h>
#include <hip/hip_bf16.h>

#define N_NODES   50000
#define N_EDGES   800000
#define HID       128
#define NGRAPH    512

typedef __attribute__((ext_vector_type(16))) _Float16 v16h;
typedef __attribute__((ext_vector_type(8)))  float    v8f;

#if __has_builtin(__builtin_amdgcn_global_load_async_to_lds_b128)
#define USE_ASYNC_LDS 1
// b128 payload type + address-space-qualified pointers the builtin expects
typedef __attribute__((__vector_size__(4 * sizeof(int)))) int b128_t;
typedef __attribute__((address_space(1))) b128_t* gptr_b128;
typedef __attribute__((address_space(3))) b128_t* lptr_b128;
#else
#define USE_ASYNC_LDS 0
#endif

__device__ __forceinline__ void wait_async_zero() {
#if __has_builtin(__builtin_amdgcn_s_wait_asynccnt)
    __builtin_amdgcn_s_wait_asynccnt(0);
#else
    asm volatile("s_wait_asynccnt 0x0" ::: "memory");
#endif
}

// ---------------------------------------------------------------------------
// C = act(act_in(A) @ W + bias) via v_wmma_f32_16x16x32_f16 (wave32).
// A: [M x K] f32 row-major, W: [K x NN] f32 row-major, C: [M x NN] f32.
// K, NN compile-time (K in {64,128}, NN in {64,128}); M % 16 == 0.
// Flags are template bools -> fully specialized, branch-free epilogue.
// Block = 256 threads (8 waves); block computes BM=2 M-tiles (32 rows) x NN.
// W staged TRANSPOSED in LDS as f16 (+8-half pad: lane stride 68 dwords -> 4
// banks apart, conflict-free 32B B reads). A tile DMA'd raw to LDS with
// global_load_async_to_lds_b128 (ASYNCcnt), converted once to f16 (+ReLU).
// Per wave: nt = wave % (NN/16) fixed -> B fragments cached in VGPRs and
// reused across M-tiles.
// WMMA VGPR layouts per CDNA5 ISA 7.12.2:
//   A f16 16x32: h=lane/16, m=lane%16; elem e -> K = e+16h (code's own order; ISA table differs)
//   B f16 32x16: elem e -> K = e+16h (code's own order), N = m
//   C/D f32    : elem v -> M = v+8h,  N = m
// ---------------------------------------------------------------------------
template<int K, int NN, bool RELU_A, bool HAS_BIAS, bool RELU_OUT>
__global__ __launch_bounds__(256) void gemm_wmma_f16(
    const float* __restrict__ A, const float* __restrict__ W,
    const float* __restrict__ bias, float* __restrict__ C, int mTiles)
{
    constexpr int BM  = 2;        // M-tiles per block
    constexpr int KP  = K + 8;    // f16 stride (halves)
    constexpr int KPF = K + 4;    // f32 stride (dwords)
    constexpr int NCH = K / 32;   // 32-wide K chunks
    constexpr int NT  = NN / 16;  // N tiles

    __shared__ __align__(16) _Float16 sW[NN * KP];       // [n][k] transposed
    __shared__ __align__(16) _Float16 sA[BM * 16 * KP];  // [r][k] f16
#if USE_ASYNC_LDS
    __shared__ __align__(16) float    sAraw[BM * 16 * KPF];
#endif

    const int tid  = threadIdx.x;
    const int mt0  = blockIdx.x * BM;
    const int mtb  = (mTiles - mt0 < BM) ? (mTiles - mt0) : BM;
    const int rows = mtb * 16;
    const int row0 = mt0 * 16;

#if USE_ASYNC_LDS
    // Kick off async DMA of the raw f32 A tile into LDS (16B per lane-op).
    constexpr int KC4 = K / 4;
    for (int idx4 = tid; idx4 < rows * KC4; idx4 += 256) {
        int r = idx4 / KC4, c4 = idx4 - r * KC4;
        __builtin_amdgcn_global_load_async_to_lds_b128(
            (gptr_b128)(void*)(A + (size_t)(row0 + r) * K + (c4 << 2)),
            (lptr_b128)(void*)(&sAraw[r * KPF + (c4 << 2)]),
            0, 0);
    }
#endif

    // Stage W transposed (f32 -> f16), overlapped with the async A DMA.
    for (int idx = tid; idx < K * NN; idx += 256) {
        int k = idx / NN, n = idx - k * NN;
        sW[n * KP + k] = (_Float16)W[idx];
    }

#if USE_ASYNC_LDS
    wait_async_zero();
    __syncthreads();
    // Convert A tile f32 -> f16 once (fused optional ReLU).
    for (int idx = tid; idx < rows * K; idx += 256) {
        int r = idx / K, k = idx - r * K;
        float v = sAraw[r * KPF + k];
        if (RELU_A) v = fmaxf(v, 0.0f);
        sA[r * KP + k] = (_Float16)v;
    }
#else
    for (int idx = tid; idx < rows * K; idx += 256) {
        int r = idx / K, k = idx - r * K;
        float v = A[(size_t)(row0 + r) * K + k];
        if (RELU_A) v = fmaxf(v, 0.0f);
        sA[r * KP + k] = (_Float16)v;
    }
#endif
    __syncthreads();

    const int wave = tid >> 5;
    const int lane = tid & 31;
    const int half = lane >> 4;
    const int m    = lane & 15;

    const int nt      = wave % NT;   // fixed per wave -> B regs reusable
    const int mtStart = wave / NT;
    const int mtStep  = 8 / NT;
    const int n0      = nt << 4;

    // Hoisted, loop-invariant bias value for this wave's output column.
    const float bv = HAS_BIAS ? bias[n0 + m] : 0.0f;

    // Cache this wave's B fragments for all K chunks in VGPRs.
    v16h bf[NCH];
#pragma unroll
    for (int kc = 0; kc < NCH; ++kc) {
#pragma unroll
        for (int e = 0; e < 16; ++e)
            bf[kc][e] = sW[(n0 + m) * KP + (kc << 5) + (half << 4) + e];
    }

    for (int mt = mtStart; mt < mtb; mt += mtStep) {
        v8f c = {};
#pragma unroll
        for (int kc = 0; kc < NCH; ++kc) {
            v16h a;
#pragma unroll
            for (int e = 0; e < 16; ++e)
                a[e] = sA[(mt * 16 + m) * KP + (kc << 5) +
                          (half << 4) + e];
            c = __builtin_amdgcn_wmma_f32_16x16x32_f16(
                    false, a, false, bf[kc], (short)0, c, false, false);
        }
        // Branch-free epilogue: one base pointer, compile-time row offsets.
        float* Cp = C + (size_t)(row0 + mt * 16 + (half << 3)) * NN + n0 + m;
#pragma unroll
        for (int v = 0; v < 8; ++v) {
            float val = c[v] + bv;
            if (RELU_OUT) val = fmaxf(val, 0.0f);
            Cp[(size_t)v * NN] = val;
        }
    }
}

// --------------------------- small helper kernels ---------------------------

__global__ void zero_f32(float* __restrict__ p, int n) {
    int i = blockIdx.x * 256 + threadIdx.x;
    if (i < n) p[i] = 0.0f;
}

__global__ void deg_count(const long long* __restrict__ dst,
                          float* __restrict__ deg, int E) {
    int e = blockIdx.x * 256 + threadIdx.x;
    if (e < E) atomicAdd(&deg[(int)dst[e]], 1.0f);
}

// deg -> 1/sqrt(deg + 1)   (+1 = self loop; always > 0)
__global__ void finalize_dinv(float* __restrict__ deg, int n) {
    int i = blockIdx.x * 256 + threadIdx.x;
    if (i < n) deg[i] = rsqrtf(deg[i] + 1.0f);
}

// agg[i, :] = bias + dinv[i]^2 * h[i, :]  (self-loop message + bias init)
__global__ void agg_init(const float* __restrict__ h,
                         const float* __restrict__ dinv,
                         const float* __restrict__ bias,
                         float* __restrict__ agg, int total4) {
    int i = blockIdx.x * 256 + threadIdx.x;   // index in float4 units
    if (i >= total4) return;
    int node = i >> 5, c4 = i & 31;
    float di = dinv[node];
    float s = di * di;
    float4 hv = ((const float4*)h)[i];
    float4 bv = ((const float4*)bias)[c4];
    float4 o;
    o.x = bv.x + s * hv.x;  o.y = bv.y + s * hv.y;
    o.z = bv.z + s * hv.z;  o.w = bv.w + s * hv.w;
    ((float4*)agg)[i] = o;
}

// one wave per edge: agg[dst, :] += dinv[src]*dinv[dst] * h[src, :]
__global__ __launch_bounds__(256) void edge_scatter(
    const long long* __restrict__ src, const long long* __restrict__ dst,
    const float* __restrict__ dinv, const float* __restrict__ h,
    float* __restrict__ agg, int E)
{
    int e = blockIdx.x * 8 + (threadIdx.x >> 5);
    if (e >= E) return;
    int lane = threadIdx.x & 31;
    int s = (int)src[e], d = (int)dst[e];
    float w = dinv[s] * dinv[d];
    float4 hv = ((const float4*)(h + (size_t)s * HID))[lane];  // 128B/lane-grp
    float* ad = agg + (size_t)d * HID + (lane << 2);
    atomicAdd(ad + 0, w * hv.x);
    atomicAdd(ad + 1, w * hv.y);
    atomicAdd(ad + 2, w * hv.z);
    atomicAdd(ad + 3, w * hv.w);
}

// one wave per node: sums[batch[i], :] += relu(agg[i, :]); cnt[batch[i]] += 1
__global__ __launch_bounds__(256) void pool_nodes(
    const float* __restrict__ agg, const long long* __restrict__ batch,
    float* __restrict__ sums, float* __restrict__ cnt, int n)
{
    int node = blockIdx.x * 8 + (threadIdx.x >> 5);
    if (node >= n) return;
    int lane = threadIdx.x & 31;
    int g = (int)batch[node];
    float4 hv = ((const float4*)(agg + (size_t)node * HID))[lane];
    float* sg = sums + g * HID + (lane << 2);
    atomicAdd(sg + 0, fmaxf(hv.x, 0.0f));
    atomicAdd(sg + 1, fmaxf(hv.y, 0.0f));
    atomicAdd(sg + 2, fmaxf(hv.z, 0.0f));
    atomicAdd(sg + 3, fmaxf(hv.w, 0.0f));
    if (lane == 0) atomicAdd(&cnt[g], 1.0f);
}

// in-place: sums[g, :] /= max(cnt[g], 1)
__global__ void gmean(float* __restrict__ sums, const float* __restrict__ cnt,
                      int total) {
    int i = blockIdx.x * 256 + threadIdx.x;
    if (i < total) sums[i] = sums[i] / fmaxf(cnt[i >> 7], 1.0f);
}

// out[g] = emb[g, :] . w + b  (one wave per graph, shuffle reduction)
__global__ __launch_bounds__(256) void fout_proj(
    const float* __restrict__ emb, const float* __restrict__ w,
    const float* __restrict__ b, float* __restrict__ out, int G)
{
    int g = blockIdx.x * 8 + (threadIdx.x >> 5);
    if (g >= G) return;
    int lane = threadIdx.x & 31;
    float4 ev = ((const float4*)(emb + (size_t)g * HID))[lane];
    float4 wv = ((const float4*)w)[lane];
    float acc = ev.x * wv.x + ev.y * wv.y + ev.z * wv.z + ev.w * wv.w;
    for (int off = 16; off > 0; off >>= 1) acc += __shfl_down(acc, off, 32);
    if (lane == 0) out[g] = acc + b[0];
}

// ---------------------------------------------------------------------------

extern "C" void kernel_launch(void* const* d_in, const int* in_sizes, int n_in,
                              void* d_out, int out_size, void* d_ws, size_t ws_size,
                              hipStream_t stream) {
    const float*     x      = (const float*)d_in[0];
    const long long* ei     = (const long long*)d_in[1];   // [2, E] int64
    const long long* batch  = (const long long*)d_in[2];   // [N] int64
    const float*     w1     = (const float*)d_in[3];
    const float*     b1     = (const float*)d_in[4];
    const float*     w2     = (const float*)d_in[5];
    const float*     b2     = (const float*)d_in[6];
    const float*     fc1_w  = (const float*)d_in[7];
    const float*     fc1_b  = (const float*)d_in[8];
    const float*     femb_w = (const float*)d_in[9];
    const float*     femb_b = (const float*)d_in[10];
    const float*     fout_w = (const float*)d_in[11];
    const float*     fout_b = (const float*)d_in[12];

    float* emb_out = (float*)d_out;                 // [512, 128]
    float* sc_out  = emb_out + NGRAPH * HID;        // [512, 1]

    const long long* esrc = ei;
    const long long* edst = ei + N_EDGES;

    // workspace layout (floats)
    float* ws   = (float*)d_ws;
    float* hA   = ws;                               // N*128
    float* hB   = hA + (size_t)N_NODES * HID;       // N*128
    float* dinv = hB + (size_t)N_NODES * HID;       // N (deg then rsqrt)
    float* sums = dinv + N_NODES;                   // 512*128 (then g in place)
    float* cnt  = sums + NGRAPH * HID;              // 512
    float* t1   = cnt + NGRAPH;                     // 512*64

    const int mTiles   = N_NODES / 16;              // 3125
    const int gTiles   = NGRAPH / 16;               // 32
    const int convBlks = (mTiles + 1) / 2;          // 1563
    const int mlpBlks  = (gTiles + 1) / 2;          // 16

    // degree / normalization
    zero_f32<<<(N_NODES + 255) / 256, 256, 0, stream>>>(dinv, N_NODES);
    deg_count<<<(N_EDGES + 255) / 256, 256, 0, stream>>>(edst, dinv, N_EDGES);
    finalize_dinv<<<(N_NODES + 255) / 256, 256, 0, stream>>>(dinv, N_NODES);

    // layer 1: h1 = x @ w1 ; agg1 = scatter + selfloop + b1
    gemm_wmma_f16<HID, HID, false, false, false><<<convBlks, 256, 0, stream>>>(
        x, w1, nullptr, hA, mTiles);
    agg_init<<<(N_NODES * 32 + 255) / 256, 256, 0, stream>>>(
        hA, dinv, b1, hB, N_NODES * 32);
    edge_scatter<<<N_EDGES / 8, 256, 0, stream>>>(esrc, edst, dinv, hA, hB, N_EDGES);

    // layer 2: h2 = relu(agg1) @ w2 ; agg2 = scatter + selfloop + b2
    gemm_wmma_f16<HID, HID, true, false, false><<<convBlks, 256, 0, stream>>>(
        hB, w2, nullptr, hA, mTiles);
    agg_init<<<(N_NODES * 32 + 255) / 256, 256, 0, stream>>>(
        hA, dinv, b2, hB, N_NODES * 32);
    edge_scatter<<<N_EDGES / 8, 256, 0, stream>>>(esrc, edst, dinv, hA, hB, N_EDGES);

    // global mean pool over relu(agg2)
    zero_f32<<<(NGRAPH * (HID + 1) + 255) / 256, 256, 0, stream>>>(
        sums, NGRAPH * (HID + 1));
    pool_nodes<<<N_NODES / 8, 256, 0, stream>>>(hB, batch, sums, cnt, N_NODES);
    gmean<<<(NGRAPH * HID) / 256, 256, 0, stream>>>(sums, cnt, NGRAPH * HID);

    // MLP head
    gemm_wmma_f16<HID, HID / 2, false, true, true><<<mlpBlks, 256, 0, stream>>>(
        sums, fc1_w, fc1_b, t1, gTiles);
    gemm_wmma_f16<HID / 2, HID, false, true, true><<<mlpBlks, 256, 0, stream>>>(
        t1, femb_w, femb_b, emb_out, gTiles);
    fout_proj<<<NGRAPH / 8, 256, 0, stream>>>(emb_out, fout_w, fout_b, sc_out,
                                              NGRAPH);
}
